// VardaGPTAssociative_39694087750292
// MI455X (gfx1250) — matrix operand order repaired and checked by reading, not compiled, not run on hardware
//
#include <hip/hip_runtime.h>
#include <math.h>

// ---------------- constants (from reference) ----------------
#define CB   2
#define CT   512
#define CD   768
#define CM   10000
#define CK   5
#define CL   12
#define CH   12
#define CHD  64
#define CFF  3072
#define CV   50257

// ---------------- WMMA types ----------------
typedef __attribute__((ext_vector_type(16))) __bf16 v16bf;
typedef __attribute__((ext_vector_type(8)))  float  v8f;

union Frag {
  v16bf v;
  uint4 q[2];
};

__device__ __forceinline__ unsigned short f2bf(float f) {
  unsigned u = __float_as_uint(f);
  u += 0x7fffu + ((u >> 16) & 1u);          // round-to-nearest-even
  return (unsigned short)(u >> 16);
}
__device__ __forceinline__ unsigned pack2(float x, float y) {
  return (unsigned)f2bf(x) | ((unsigned)f2bf(y) << 16);
}

// =====================================================================
// Generic GEMM: C[M x N] = alpha * A[M x K] * B + bscale*bias[n] (+gelu)(+res)
// A fp32 row-major. BT=true: B is [N][K] row-major (kNN memory, wte).
//                   BT=false: B is [K][N] row-major (layer weights).
// Block: 256 threads = 8 waves (2x4), tile 64(M) x 128(N), BK=32.
// Software pipeline: register prefetch of tile k+1 issued before WMMAs of
// tile k; double-buffered LDS; global_prefetch_b8 two tiles ahead.
// =====================================================================
template <bool BT>
__global__ __launch_bounds__(256) void gemm_bf16_wmma(
    const float* __restrict__ A, long long asA,
    const float* __restrict__ Bm, long long asB,
    float* __restrict__ C, long long asC,
    const float* __restrict__ bias, long long asBias, float bscale,
    const float* __restrict__ res,
    int Ndim, int Kdim, int lda, int ldb, int ldc,
    float alpha, int dogelu)
{
  __shared__ alignas(16) unsigned short As[2][64 * 40];
  __shared__ alignas(16) unsigned short Bs[2][128 * 40];

  const int bz = blockIdx.z;
  A  += (long long)bz * asA;
  Bm += (long long)bz * asB;
  C  += (long long)bz * asC;
  const float* biasp = bias ? bias + (long long)bz * asBias : nullptr;

  const int n0 = blockIdx.x * 128;
  const int m0 = blockIdx.y * 64;
  const int tid  = threadIdx.x;
  const int wave = tid >> 5, lane = tid & 31;
  const int half = lane >> 4, l16 = lane & 15;
  const int wm = wave >> 2, wn = wave & 3;   // 2 (M) x 4 (N) waves

  const v8f vzero = {0.f,0.f,0.f,0.f,0.f,0.f,0.f,0.f};
  v8f acc[2][2] = {{vzero, vzero}, {vzero, vzero}};

  const int nk = Kdim >> 5;
  float4 ra[2], rb[4];

  auto load_tiles = [&](int kt) {
    const int k0 = kt * 32;
#pragma unroll
    for (int i = 0; i < 2; ++i) {
      int e = tid + i * 256;
      int r = e >> 3, c4 = e & 7;
      ra[i] = *(const float4*)(A + (long long)(m0 + r) * lda + k0 + c4 * 4);
    }
    if (BT) {
#pragma unroll
      for (int i = 0; i < 4; ++i) {
        int e = tid + i * 256;
        int r = e >> 3, c4 = e & 7;
        rb[i] = make_float4(0.f, 0.f, 0.f, 0.f);
        if (n0 + r < Ndim)
          rb[i] = *(const float4*)(Bm + (long long)(n0 + r) * ldb + k0 + c4 * 4);
      }
    } else {
#pragma unroll
      for (int i = 0; i < 4; ++i) {
        int e = tid + i * 256;
        int kr = e >> 5, c4 = e & 31;
        rb[i] = *(const float4*)(Bm + (long long)(k0 + kr) * ldb + n0 + c4 * 4);
      }
    }
  };

  auto store_tiles = [&](int cur) {
#pragma unroll
    for (int i = 0; i < 2; ++i) {
      int e = tid + i * 256;
      int r = e >> 3, c4 = e & 7;
      uint2 p; p.x = pack2(ra[i].x, ra[i].y); p.y = pack2(ra[i].z, ra[i].w);
      *(uint2*)(&As[cur][r * 40 + c4 * 4]) = p;
    }
    if (BT) {
#pragma unroll
      for (int i = 0; i < 4; ++i) {
        int e = tid + i * 256;
        int r = e >> 3, c4 = e & 7;
        uint2 p; p.x = pack2(rb[i].x, rb[i].y); p.y = pack2(rb[i].z, rb[i].w);
        *(uint2*)(&Bs[cur][r * 40 + c4 * 4]) = p;
      }
    } else {
#pragma unroll
      for (int i = 0; i < 4; ++i) {
        int e = tid + i * 256;
        int kr = e >> 5, c4 = e & 31;
        Bs[cur][(c4 * 4 + 0) * 40 + kr] = f2bf(rb[i].x);
        Bs[cur][(c4 * 4 + 1) * 40 + kr] = f2bf(rb[i].y);
        Bs[cur][(c4 * 4 + 2) * 40 + kr] = f2bf(rb[i].z);
        Bs[cur][(c4 * 4 + 3) * 40 + kr] = f2bf(rb[i].w);
      }
    }
  };

  load_tiles(0);
  for (int kt = 0; kt < nk; ++kt) {
    const int cur = kt & 1;
    store_tiles(cur);
    __syncthreads();                         // waits dscnt: safe double buffer

    // L2 prefetch two tiles ahead (global_prefetch_b8)
    if (kt + 2 < nk) {
      const int k2 = (kt + 2) * 32;
      int r = tid >> 3, c4 = tid & 7;
      __builtin_prefetch(A + (long long)(m0 + r) * lda + k2 + c4 * 4, 0, 1);
      if (BT) {
        if (n0 + r < Ndim)
          __builtin_prefetch(Bm + (long long)(n0 + r) * ldb + k2 + c4 * 4, 0, 1);
      } else {
        int kr = tid >> 5, c42 = tid & 31;
        __builtin_prefetch(Bm + (long long)(k2 + kr) * ldb + n0 + c42 * 4, 0, 1);
      }
    }
    // register prefetch of next tile, issued before this tile's WMMAs
    if (kt + 1 < nk) load_tiles(kt + 1);

    // ---- fragment loads (ISA 16-bit A/B VGPR layouts) ----
    Frag a[2], b[2];
#pragma unroll
    for (int mi = 0; mi < 2; ++mi) {
      const unsigned short* p = &As[cur][(wm * 32 + mi * 16 + l16) * 40 + half * 8];
      a[mi].q[0] = *(const uint4*)p;        // K 0..7  (or 8..15 upper half)
      a[mi].q[1] = *(const uint4*)(p + 16); // K 16..23 (or 24..31)
    }
#pragma unroll
    for (int ni = 0; ni < 2; ++ni) {
      const unsigned short* p = &Bs[cur][(wn * 32 + ni * 16 + l16) * 40 + half * 8];
      b[ni].q[0] = *(const uint4*)p;        // K 0..7  (or 8..15 upper half)
      b[ni].q[1] = *(const uint4*)(p + 16); // K 16..23 (or 24..31)
    }
#pragma unroll
    for (int mi = 0; mi < 2; ++mi)
#pragma unroll
      for (int ni = 0; ni < 2; ++ni)
        acc[mi][ni] = __builtin_amdgcn_wmma_f32_16x16x32_bf16(
            false, a[mi].v, false, b[ni].v, (short)0, acc[mi][ni], false, false);
  }

  // ---- epilogue: D layout row = r + 8*half, col = lane%16 ----
#pragma unroll
  for (int ni = 0; ni < 2; ++ni) {
    int col = n0 + wn * 32 + ni * 16 + l16;
    if (col >= Ndim) continue;
    float bc = biasp ? bscale * biasp[col] : 0.f;
#pragma unroll
    for (int mi = 0; mi < 2; ++mi) {
#pragma unroll
      for (int r = 0; r < 8; ++r) {
        int row = m0 + wm * 32 + mi * 16 + half * 8 + r;
        float v = alpha * acc[mi][ni][r] + bc;
        if (dogelu) {
          float t = 0.7978845608028654f * (v + 0.044715f * v * v * v);
          v = 0.5f * v * (1.f + tanhf(t));
        }
        long long ci = (long long)row * ldc + col;
        if (res) v += res[ci];
        C[ci] = v;
      }
    }
  }
}

// =====================================================================
// Flash attention (causal), WMMA for QK^T and P*V.
// Grid: (T/64, H, B), block = 128 threads = 4 waves; wave w owns q rows
// [bq + 16w, bq + 16w + 15]. K chunk = 32 keys, double-buffered + register
// prefetch of the next chunk.
// =====================================================================
__global__ __launch_bounds__(128) void flash_attn_wmma(
    const float* __restrict__ qkv, float* __restrict__ out, float scale)
{
  const int D3 = 3 * CD;
  __shared__ alignas(16) unsigned short Qs[64 * 72];
  __shared__ alignas(16) unsigned short Ks[2][32 * 72];
  __shared__ alignas(16) unsigned short Vs[2][64 * 40];
  __shared__ alignas(16) unsigned short Ps[4][16 * 40];

  const int bq = blockIdx.x * 64;
  const int h  = blockIdx.y;
  const int b  = blockIdx.z;
  const int tid  = threadIdx.x;
  const int wave = tid >> 5, lane = tid & 31;
  const int half = lane >> 4, l16 = lane & 15;
  const float* base = qkv + (long long)b * CT * D3;

  // ---- stage Q (pre-scaled) 64x64 ----
#pragma unroll
  for (int i = 0; i < 8; ++i) {
    int e = tid + i * 128;
    int r = e >> 4, c4 = e & 15;
    float4 f = *(const float4*)(base + (long long)(bq + r) * D3 + h * CHD + c4 * 4);
    uint2 p; p.x = pack2(f.x * scale, f.y * scale); p.y = pack2(f.z * scale, f.w * scale);
    *(uint2*)(&Qs[r * 72 + c4 * 4]) = p;
  }
  __syncthreads();
  Frag Aq[2];
#pragma unroll
  for (int kk = 0; kk < 2; ++kk) {
    const unsigned short* p = &Qs[(wave * 16 + l16) * 72 + kk * 32 + half * 8];
    Aq[kk].q[0] = *(const uint4*)p;
    Aq[kk].q[1] = *(const uint4*)(p + 16);
  }

  const v8f vzero = {0.f,0.f,0.f,0.f,0.f,0.f,0.f,0.f};
  float mrow[8], lrow[8];
  v8f O[4] = {vzero, vzero, vzero, vzero};
#pragma unroll
  for (int r = 0; r < 8; ++r) { mrow[r] = -1e30f; lrow[r] = 0.f; }

  float4 rk[4], rv[4];
  auto load_kv = [&](int c) {
    const int kb = c * 32;
#pragma unroll
    for (int i = 0; i < 4; ++i) {
      int e = tid + i * 128;
      int r = e >> 4, c4 = e & 15;
      rk[i] = *(const float4*)(base + (long long)(kb + r) * D3 + CD + h * CHD + c4 * 4);
      rv[i] = *(const float4*)(base + (long long)(kb + r) * D3 + 2 * CD + h * CHD + c4 * 4);
    }
  };

  const int nch = bq / 32 + 2;               // chunks with base <= max q of block
  load_kv(0);
  for (int c = 0; c < nch; ++c) {
    const int kb = c * 32;
    const int cur = c & 1;
    // ---- store prefetched K chunk [key][hd] and V chunk transposed ----
#pragma unroll
    for (int i = 0; i < 4; ++i) {
      int e = tid + i * 128;
      int r = e >> 4, c4 = e & 15;
      uint2 p; p.x = pack2(rk[i].x, rk[i].y); p.y = pack2(rk[i].z, rk[i].w);
      *(uint2*)(&Ks[cur][r * 72 + c4 * 4]) = p;
      Vs[cur][(c4 * 4 + 0) * 40 + r] = f2bf(rv[i].x);
      Vs[cur][(c4 * 4 + 1) * 40 + r] = f2bf(rv[i].y);
      Vs[cur][(c4 * 4 + 2) * 40 + r] = f2bf(rv[i].z);
      Vs[cur][(c4 * 4 + 3) * 40 + r] = f2bf(rv[i].w);
    }
    __syncthreads();                         // waits dscnt: safe double buffer
    if (c + 1 < nch) load_kv(c + 1);         // overlap next chunk with compute

    // ---- S = Q * K^T  (two 16-key subtiles, two 32-wide K steps over HD) ----
    v8f S[2] = {vzero, vzero};
#pragma unroll
    for (int ni = 0; ni < 2; ++ni) {
#pragma unroll
      for (int kk = 0; kk < 2; ++kk) {
        Frag Bk;
        const unsigned short* p = &Ks[cur][(ni * 16 + l16) * 72 + kk * 32 + half * 8];
        Bk.q[0] = *(const uint4*)p;
        Bk.q[1] = *(const uint4*)(p + 16);
        S[ni] = __builtin_amdgcn_wmma_f32_16x16x32_bf16(
            false, Aq[kk].v, false, Bk.v, (short)0, S[ni], false, false);
      }
    }

    // ---- causal mask + streaming softmax stats (per-row over 16 lanes) ----
    const int q0 = bq + wave * 16 + half * 8;
    float sc[8];
#pragma unroll
    for (int r = 0; r < 8; ++r) {
      int qg = q0 + r;
      if (kb + l16 > qg)      S[0][r] = -1e30f;
      if (kb + 16 + l16 > qg) S[1][r] = -1e30f;
      float cm = fmaxf(S[0][r], S[1][r]);
      cm = fmaxf(cm, __shfl_xor(cm, 1, 32));
      cm = fmaxf(cm, __shfl_xor(cm, 2, 32));
      cm = fmaxf(cm, __shfl_xor(cm, 4, 32));
      cm = fmaxf(cm, __shfl_xor(cm, 8, 32));
      float mn = fmaxf(mrow[r], cm);
      sc[r] = __expf(mrow[r] - mn);
      mrow[r] = mn;
      float p0 = __expf(S[0][r] - mn);
      float p1 = __expf(S[1][r] - mn);
      S[0][r] = p0; S[1][r] = p1;
      float ps = p0 + p1;
      ps += __shfl_xor(ps, 1, 32);
      ps += __shfl_xor(ps, 2, 32);
      ps += __shfl_xor(ps, 4, 32);
      ps += __shfl_xor(ps, 8, 32);
      lrow[r] = lrow[r] * sc[r] + ps;
    }

    // ---- P (D layout) -> per-wave LDS -> A fragment layout ----
#pragma unroll
    for (int r = 0; r < 8; ++r) {
      Ps[wave][(half * 8 + r) * 40 + l16]      = f2bf(S[0][r]);
      Ps[wave][(half * 8 + r) * 40 + 16 + l16] = f2bf(S[1][r]);
    }
    // rescale running O
#pragma unroll
    for (int os = 0; os < 4; ++os)
#pragma unroll
      for (int r = 0; r < 8; ++r) O[os][r] *= sc[r];

    Frag Ap;
    const unsigned short* pp = &Ps[wave][l16 * 40 + half * 8];
    Ap.q[0] = *(const uint4*)pp;
    Ap.q[1] = *(const uint4*)(pp + 16);

    // ---- O += P * V ----
#pragma unroll
    for (int os = 0; os < 4; ++os) {
      Frag Bv;
      const unsigned short* p = &Vs[cur][(os * 16 + l16) * 40 + half * 8];
      Bv.q[0] = *(const uint4*)p;
      Bv.q[1] = *(const uint4*)(p + 16);
      O[os] = __builtin_amdgcn_wmma_f32_16x16x32_bf16(
          false, Ap.v, false, Bv.v, (short)0, O[os], false, false);
    }
  }

  // ---- normalize + store ----
#pragma unroll
  for (int os = 0; os < 4; ++os)
#pragma unroll
    for (int r = 0; r < 8; ++r) {
      int row = bq + wave * 16 + half * 8 + r;
      float v = O[os][r] / lrow[r];
      out[((long long)b * CT + row) * CD + h * CHD + os * 16 + l16] = v;
    }
}

// =====================================================================
// ||m||^2 per memory row: 8 waves/block, one row per wave.
// =====================================================================
__global__ __launch_bounds__(256) void msq_k(const float* __restrict__ mem,
                                             float* __restrict__ msq)
{
  int wave = threadIdx.x >> 5, lane = threadIdx.x & 31;
  int row = blockIdx.x * 8 + wave;
  if (row >= CB * CM) return;
  const float* mr = mem + (long long)row * CD;
  float s = 0.f;
  for (int i = lane; i < CD; i += 32) { float v = mr[i]; s += v * v; }
  s += __shfl_xor(s, 1, 32);
  s += __shfl_xor(s, 2, 32);
  s += __shfl_xor(s, 4, 32);
  s += __shfl_xor(s, 8, 32);
  s += __shfl_xor(s, 16, 32);
  if (lane == 0) msq[row] = s;
}

// =====================================================================
// top-5 argmax per (b,t) over M=10000 scores + gather concat(x, retrieved).
// =====================================================================
__global__ __launch_bounds__(256) void topk_gather_k(
    float* __restrict__ scores, const float* __restrict__ memory,
    const float* __restrict__ inp, float* __restrict__ xcat)
{
  __shared__ float sv[256];
  __shared__ int   si[256];
  __shared__ int   picked[CK];
  const int bt = blockIdx.x;
  const int b  = bt / CT;
  float* row = scores + (long long)bt * CM;
  const int tid = threadIdx.x;

  for (int kk = 0; kk < CK; ++kk) {
    float best = -3.0e38f; int bi = 0;
    for (int m = tid; m < CM; m += 256) {
      float v = row[m];
      if (v > best) { best = v; bi = m; }
    }
    sv[tid] = best; si[tid] = bi;
    __syncthreads();
    for (int st = 128; st > 0; st >>= 1) {
      if (tid < st && sv[tid + st] > sv[tid]) { sv[tid] = sv[tid + st]; si[tid] = si[tid + st]; }
      __syncthreads();
    }
    if (tid == 0) { picked[kk] = si[0]; row[si[0]] = -3.0e38f; }
    __syncthreads();
  }
  const float* iv = inp + (long long)bt * CD;
  float* xr = xcat + (long long)bt * (CD + CK * CD);
  for (int i = tid; i < CD; i += 256) xr[i] = iv[i];
  for (int kk = 0; kk < CK; ++kk) {
    const float* mr = memory + ((long long)b * CM + picked[kk]) * CD;
    for (int i = tid; i < CD; i += 256) xr[CD + kk * CD + i] = mr[i];
  }
}

// =====================================================================
// LayerNorm: one 256-thread block per row of 768.
// =====================================================================
__global__ __launch_bounds__(256) void layernorm_k(
    const float* __restrict__ x, const float* __restrict__ g,
    const float* __restrict__ bta, float* __restrict__ y)
{
  __shared__ float red[256];
  const long long row = blockIdx.x;
  const float* xr = x + row * CD;
  float v[3];
  float s = 0.f;
#pragma unroll
  for (int i = 0; i < 3; ++i) { v[i] = xr[threadIdx.x + i * 256]; s += v[i]; }
  red[threadIdx.x] = s; __syncthreads();
  for (int st = 128; st > 0; st >>= 1) {
    if (threadIdx.x < st) red[threadIdx.x] += red[threadIdx.x + st];
    __syncthreads();
  }
  float mu = red[0] * (1.f / CD);
  __syncthreads();
  float ss = 0.f;
#pragma unroll
  for (int i = 0; i < 3; ++i) { float d = v[i] - mu; ss += d * d; }
  red[threadIdx.x] = ss; __syncthreads();
  for (int st = 128; st > 0; st >>= 1) {
    if (threadIdx.x < st) red[threadIdx.x] += red[threadIdx.x + st];
    __syncthreads();
  }
  float inv = rsqrtf(red[0] * (1.f / CD) + 1e-5f);
#pragma unroll
  for (int i = 0; i < 3; ++i) {
    int cidx = threadIdx.x + i * 256;
    y[row * CD + cidx] = (v[i] - mu) * inv * g[cidx] + bta[cidx];
  }
}

// =====================================================================
// h += wpe[t] broadcast over batch.
// =====================================================================
__global__ void add_wpe_k(float* __restrict__ h, const float* __restrict__ wpe)
{
  long long i = (long long)blockIdx.x * blockDim.x + threadIdx.x;
  if (i < (long long)CB * CT * CD) {
    int t = (int)((i / CD) % CT);
    int d = (int)(i % CD);
    h[i] += wpe[t * CD + d];
  }
}

// =====================================================================
// Orchestration
// =====================================================================
extern "C" void kernel_launch(void* const* d_in, const int* in_sizes, int n_in,
                              void* d_out, int out_size, void* d_ws, size_t ws_size,
                              hipStream_t stream) {
  (void)in_sizes; (void)n_in; (void)out_size; (void)ws_size;
  const float* input_vectors = (const float*)d_in[0];
  const float* memory     = (const float*)d_in[1];
  const float* fc_w       = (const float*)d_in[2];
  const float* fc_b       = (const float*)d_in[3];
  const float* wpe        = (const float*)d_in[4];
  const float* wte        = (const float*)d_in[5];
  const float* ln1_g      = (const float*)d_in[6];
  const float* ln1_b      = (const float*)d_in[7];
  const float* attn_w     = (const float*)d_in[8];
  const float* attn_b     = (const float*)d_in[9];
  const float* proj_w     = (const float*)d_in[10];
  const float* proj_b     = (const float*)d_in[11];
  const float* ln2_g      = (const float*)d_in[12];
  const float* ln2_b      = (const float*)d_in[13];
  const float* mlp_fc_w   = (const float*)d_in[14];
  const float* mlp_fc_b   = (const float*)d_in[15];
  const float* mlp_proj_w = (const float*)d_in[16];
  const float* mlp_proj_b = (const float*)d_in[17];
  const float* lnf_g      = (const float*)d_in[18];
  const float* lnf_b      = (const float*)d_in[19];
  float* logits = (float*)d_out;

  float* ws = (float*)d_ws;
  size_t off = 0;
  auto take = [&](size_t n) { float* p = ws + off; off += n; return p; };
  float* msq    = take((size_t)CB * CM);
  float* scores = take((size_t)CB * CT * CM);
  float* xcat   = take((size_t)CB * CT * (CD + CK * CD));
  float* hbuf   = take((size_t)CB * CT * CD);
  float* abuf   = take((size_t)CB * CT * CD);
  float* qkvb   = take((size_t)CB * CT * 3 * CD);
  float* attno  = take((size_t)CB * CT * CD);
  float* mlphb  = take((size_t)CB * CT * CFF);

  const int BT64 = (CB * CT) / 64;   // 16

  // 1) ||m||^2
  msq_k<<<dim3((CB * CM + 7) / 8), 256, 0, stream>>>(memory, msq);

  // 2) kNN scores = 2*Q.M^T - ||m||^2   (per-batch GEMM, B=[N][K])
  gemm_bf16_wmma<true><<<dim3((CM + 127) / 128, CT / 64, CB), 256, 0, stream>>>(
      input_vectors, (long long)CT * CD,
      memory,        (long long)CM * CD,
      scores,        (long long)CT * CM,
      msq,           (long long)CM, -1.f,
      nullptr,
      CM, CD, CD, CD, CM, 2.f, 0);

  // 3) top-5 + gather into concat buffer
  topk_gather_k<<<dim3(CB * CT), 256, 0, stream>>>(scores, memory, input_vectors, xcat);

  // 4) fc fuse: h = xcat @ fc_w + fc_b
  gemm_bf16_wmma<false><<<dim3(CD / 128, BT64, 1), 256, 0, stream>>>(
      xcat, 0, fc_w, 0, hbuf, 0, fc_b, 0, 1.f, nullptr,
      CD, CD + CK * CD, CD + CK * CD, CD, CD, 1.f, 0);

  // 5) + wpe
  add_wpe_k<<<dim3((CB * CT * CD + 255) / 256), 256, 0, stream>>>(hbuf, wpe);

  for (int l = 0; l < CL; ++l) {
    layernorm_k<<<dim3(CB * CT), 256, 0, stream>>>(hbuf, ln1_g + l * CD, ln1_b + l * CD, abuf);
    gemm_bf16_wmma<false><<<dim3(3 * CD / 128, BT64, 1), 256, 0, stream>>>(
        abuf, 0, attn_w + (size_t)l * CD * 3 * CD, 0, qkvb, 0,
        attn_b + (size_t)l * 3 * CD, 0, 1.f, nullptr,
        3 * CD, CD, CD, 3 * CD, 3 * CD, 1.f, 0);
    flash_attn_wmma<<<dim3(CT / 64, CH, CB), 128, 0, stream>>>(qkvb, attno, 0.125f);
    gemm_bf16_wmma<false><<<dim3(CD / 128, BT64, 1), 256, 0, stream>>>(
        attno, 0, proj_w + (size_t)l * CD * CD, 0, hbuf, 0,
        proj_b + (size_t)l * CD, 0, 1.f, hbuf,
        CD, CD, CD, CD, CD, 1.f, 0);
    layernorm_k<<<dim3(CB * CT), 256, 0, stream>>>(hbuf, ln2_g + l * CD, ln2_b + l * CD, abuf);
    gemm_bf16_wmma<false><<<dim3(CFF / 128, BT64, 1), 256, 0, stream>>>(
        abuf, 0, mlp_fc_w + (size_t)l * CD * CFF, 0, mlphb, 0,
        mlp_fc_b + (size_t)l * CFF, 0, 1.f, nullptr,
        CFF, CD, CD, CFF, CFF, 1.f, 1);
    gemm_bf16_wmma<false><<<dim3(CD / 128, BT64, 1), 256, 0, stream>>>(
        mlphb, 0, mlp_proj_w + (size_t)l * CFF * CD, 0, hbuf, 0,
        mlp_proj_b + (size_t)l * CD, 0, 1.f, hbuf,
        CD, CFF, CFF, CD, CD, 1.f, 0);
  }

  layernorm_k<<<dim3(CB * CT), 256, 0, stream>>>(hbuf, lnf_g, lnf_b, abuf);

  // lm_head: logits = h @ wte^T  (wte is [V][D] == [N][K])
  gemm_bf16_wmma<true><<<dim3((CV + 127) / 128, BT64, 1), 256, 0, stream>>>(
      abuf, 0, wte, 0, logits, 0, nullptr, 0, 0.f, nullptr,
      CV, CD, CD, CD, CV, 1.f, 0);
}
